// MambaBEAT_44581760532986
// MI455X (gfx1250) — matrix operand order repaired and checked by reading, not compiled, not run on hardware
//
#include <hip/hip_runtime.h>
#include <hip/hip_bf16.h>
#include <math.h>

// ---------------------------------------------------------------------------
// Mamba (2 layers) forward for gfx1250 / MI455X.
//   B=8, L=4096, IN_CH=64, EXP=128, NSTATE=16, DT_RANK=4, KSIZE=4, OUT_CH=5
// GEMMs: V_WMMA_F32_16X16X32_F16 (wave32, fp32 accumulate, K fully unrolled,
//        32x16 output per wave, fully templated shapes -> 32-bit addressing).
// Scan:  TDM tensor_load_to_lds staging of shared B/C/dt rows + register-
//        resident recurrent state.
// ---------------------------------------------------------------------------

#define NL     2
#define INC    64
#define EXPD   128
#define NST    16
#define DTR    4
#define KSZ    4
#define OUTC   5
#define BB     8
#define LL     4096
#define MROWS  (BB * LL)          // 32768, multiple of 128
#define SELN   (2 * NST + DTR)    // 36
#define SELNP  48                 // padded to 3 n-tiles

typedef __attribute__((ext_vector_type(16))) _Float16 v16h;
typedef __attribute__((ext_vector_type(8)))  float    v8f;
typedef __attribute__((ext_vector_type(4)))  unsigned int u32x4;
typedef __attribute__((ext_vector_type(8)))  int      i32x8;
typedef __attribute__((ext_vector_type(4)))  int      i32x4;

__device__ __forceinline__ float silu_f(float v) {
    return v / (1.0f + __expf(-v));
}

// A-fragment (16x32 f16, ISA 7.12.2): lane half g, elements 0-7 = K[g*8 .. g*8+7],
// elements 8-15 = K[16+g*8 .. 16+g*8+7]. p points at A_row + kb + g*8.
__device__ __forceinline__ v16h frag_a(const float* __restrict__ p) {
    const float4 x0 = *(const float4*)(p);
    const float4 x1 = *(const float4*)(p + 4);
    const float4 x2 = *(const float4*)(p + 16);
    const float4 x3 = *(const float4*)(p + 20);
    v16h f;
    f[0]  = (_Float16)x0.x; f[1]  = (_Float16)x0.y; f[2]  = (_Float16)x0.z; f[3]  = (_Float16)x0.w;
    f[4]  = (_Float16)x1.x; f[5]  = (_Float16)x1.y; f[6]  = (_Float16)x1.z; f[7]  = (_Float16)x1.w;
    f[8]  = (_Float16)x2.x; f[9]  = (_Float16)x2.y; f[10] = (_Float16)x2.z; f[11] = (_Float16)x2.w;
    f[12] = (_Float16)x3.x; f[13] = (_Float16)x3.y; f[14] = (_Float16)x3.z; f[15] = (_Float16)x3.w;
    return f;
}

// B-fragment (32x16 f16): lane half g, elements 0-7 = K[g*8 .. g*8+7], elements
// 8-15 = K[16+g*8 .. 16+g*8+7] of column n = lane&15. p points at W_row(n) + kb + g*8.
__device__ __forceinline__ v16h frag_b(const float* __restrict__ p) {
    const float4 x0 = *(const float4*)(p);
    const float4 x1 = *(const float4*)(p + 4);
    const float4 x2 = *(const float4*)(p + 16);
    const float4 x3 = *(const float4*)(p + 20);
    v16h f;
    f[0]  = (_Float16)x0.x; f[1]  = (_Float16)x0.y; f[2]  = (_Float16)x0.z; f[3]  = (_Float16)x0.w;
    f[4]  = (_Float16)x1.x; f[5]  = (_Float16)x1.y; f[6]  = (_Float16)x1.z; f[7]  = (_Float16)x1.w;
    f[8]  = (_Float16)x2.x; f[9]  = (_Float16)x2.y; f[10] = (_Float16)x2.z; f[11] = (_Float16)x2.w;
    f[12] = (_Float16)x3.x; f[13] = (_Float16)x3.y; f[14] = (_Float16)x3.z; f[15] = (_Float16)x3.w;
    return f;
}

// ---------------------------------------------------------------------------
// WMMA GEMM:  C[MROWS x NREAL] = A[MROWS x K] * W[NPAD x K]^T (+ bias)
// W padded to NPAD rows (16-multiple) so all B-loads are unguarded b128.
// Each wave computes a 32(M) x 16(N) tile: two accumulators share one B frag.
// All shape params are compile-time -> pure 32-bit address arithmetic.
// grid = (NPAD/16, MROWS/128), block = 128 (4 waves).
// ---------------------------------------------------------------------------
template <int K, int NPAD, int NREAL, bool HASBIAS>
__global__ void wmma_gemm_bias(const float* __restrict__ A,
                               const float* __restrict__ W,
                               const float* __restrict__ bias,
                               float* __restrict__ C) {
    const int wave = threadIdx.x >> 5;
    const int lane = threadIdx.x & 31;
    const int row  = lane & 15;       // M-row in tile (A) / N-col (B, C/D)
    const int g    = lane >> 4;       // lane-half selector

    const unsigned mbase = (blockIdx.y * 4 + wave) * 32;
    const unsigned ncol  = blockIdx.x * 16 + row;

    const float* __restrict__ Wrow = W + ncol * K;
    const float* __restrict__ Ar0  = A + (mbase + row) * K;
    const float* __restrict__ Ar1  = Ar0 + 16 * K;

    v8f acc0 = {}, acc1 = {};
#pragma unroll
    for (int kb = 0; kb < K; kb += 32) {
        const v16h b  = frag_b(Wrow + kb + g * 8);
        const v16h a0 = frag_a(Ar0 + kb + g * 8);
        const v16h a1 = frag_a(Ar1 + kb + g * 8);
        acc0 = __builtin_amdgcn_wmma_f32_16x16x32_f16(false, a0, false, b,
                                                      (short)0, acc0, false, false);
        acc1 = __builtin_amdgcn_wmma_f32_16x16x32_f16(false, a1, false, b,
                                                      (short)0, acc1, false, false);
    }

    if (NREAL == NPAD || ncol < (unsigned)NREAL) {
        const float bv = HASBIAS ? bias[ncol] : 0.0f;
        const unsigned cbase = (mbase + 8u * g) * NREAL + ncol;
#pragma unroll
        for (int i = 0; i < 8; ++i) {   // D: VGPR i -> M = i + 8*lanehalf
            C[cbase + (unsigned)i * NREAL]                = acc0[i] + bv;
            C[cbase + (unsigned)(16 + i) * NREAL]         = acc1[i] + bv;
        }
    }
}

// ---------------------------------------------------------------------------
// Zero-pad sel_w (36 x 128) -> (48 x 128) so the GEMM B-loads are unguarded.
// ---------------------------------------------------------------------------
__global__ void pack_selw(const float* __restrict__ selw, float* __restrict__ wpad) {
    const int idx = blockIdx.x * blockDim.x + threadIdx.x;
    if (idx >= SELNP * EXPD) return;
    const int n = idx / EXPD, k = idx % EXPD;
    wpad[idx] = (n < SELN) ? selw[n * EXPD + k] : 0.0f;
}

// ---------------------------------------------------------------------------
// Depthwise causal conv (KSIZE=4) + bias + SiLU over xz[..., :EXP]
// ---------------------------------------------------------------------------
__global__ void conv_silu_kernel(const float* __restrict__ xz,
                                 const float* __restrict__ cw,
                                 const float* __restrict__ cb,
                                 float* __restrict__ xca) {
    const size_t idx = (size_t)blockIdx.x * blockDim.x + threadIdx.x;
    if (idx >= (size_t)MROWS * EXPD) return;
    const int e = idx % EXPD;
    const int l = (idx / EXPD) % LL;
    const int b = idx / ((size_t)EXPD * LL);

    const float* base = xz + (size_t)b * LL * (2 * EXPD);
    float s = cb[e];
#pragma unroll
    for (int t = 0; t < KSZ; ++t) {
        const int ls = l - (KSZ - 1) + t;
        if (ls >= 0) s += cw[e * KSZ + t] * base[(size_t)ls * (2 * EXPD) + e];
    }
    xca[idx] = silu_f(s);
}

// ---------------------------------------------------------------------------
// Selective scan. One block per batch, one thread per channel e (128 threads,
// 4 waves). 16 recurrent states in registers. Per-timestep B/C/dt rows
// (36 floats shared by all channels) staged through LDS in 16-step chunks via
// the Tensor Data Mover when available (1-D D#, data_size=4B, tile=576 elems).
// ---------------------------------------------------------------------------
#define SCAN_CHUNK   16
#define CHUNK_ELEMS  (SCAN_CHUNK * SELN)   // 576 floats = 2304 B

#if __has_builtin(__builtin_amdgcn_tensor_load_to_lds)
#define HAVE_TDM 1
// Build a D# for a flat 1-D copy of CHUNK_ELEMS fp32 from `gp` into LDS offset 0
// (sbcd is the kernel's only LDS object, so its offset is 0).
// Bit layout per CDNA5 ISA.
__device__ __forceinline__ void tdm_load_chunk(const float* __restrict__ gp) {
    const unsigned long long ga = (unsigned long long)(uintptr_t)gp;
    u32x4 g0;
    g0[0] = 1u;                                    // count=1, user descriptor
    g0[1] = 0u;                                    // lds_addr = 0
    g0[2] = (unsigned)(ga & 0xFFFFFFFFull);        // global_addr[31:0]
    g0[3] = (unsigned)((ga >> 32) & 0x1FFFFFFull)  // global_addr[56:32]
          | (2u << 30);                            // type = 2 ("image")
    i32x8 g1;
    g1[0] = (int)(2u << 16);                       // data_size = 4 bytes
    g1[1] = (int)(((unsigned)CHUNK_ELEMS & 0xFFFFu) << 16);    // tensor_dim0 lo16
    g1[2] = (int)(((unsigned)CHUNK_ELEMS >> 16) | (1u << 16)); // dim0 hi16 | tensor_dim1=1
    g1[3] = (int)((unsigned)CHUNK_ELEMS << 16);    // tensor_dim1 hi=0 | tile_dim0=576
    g1[4] = 0;                                     // tile_dim1=0, tile_dim2=0
    g1[5] = (int)CHUNK_ELEMS;                      // tensor_dim0_stride lo32
    g1[6] = 0;                                     // stride hi | dim1_stride lo
    g1[7] = 0;
    const i32x4 z4 = {0, 0, 0, 0};
#if defined(__clang_major__) && (__clang_major__ >= 23)
    const i32x8 z8 = {0, 0, 0, 0, 0, 0, 0, 0};
    __builtin_amdgcn_tensor_load_to_lds(g0, g1, z4, z4, z8, 0);
#else
    __builtin_amdgcn_tensor_load_to_lds(g0, g1, z4, z4, 0);
#endif
}
#endif

__global__ void scan_kernel(const float* __restrict__ bcd,   // (B,L,36)
                            const float* __restrict__ xca,   // (B,L,128)
                            const float* __restrict__ xz,    // (B,L,256) -> z at +128
                            const float* __restrict__ dtw,   // (128,4)
                            const float* __restrict__ dtb,   // (128)
                            const float* __restrict__ a_log, // (128,16)
                            const float* __restrict__ dpar,  // (128)
                            float* __restrict__ yg) {        // (B,L,128)
    __shared__ float sbcd[CHUNK_ELEMS];            // only LDS object (offset 0)

    const int e = threadIdx.x;       // 0..127
    const int b = blockIdx.x;        // 0..7

    float Arow[NST];
#pragma unroll
    for (int n = 0; n < NST; ++n) Arow[n] = -__expf(a_log[e * NST + n]);
    const float w0 = dtw[e * DTR + 0], w1 = dtw[e * DTR + 1];
    const float w2 = dtw[e * DTR + 2], w3 = dtw[e * DTR + 3];
    const float dbias = dtb[e];
    const float dd    = dpar[e];

    float h[NST];
#pragma unroll
    for (int n = 0; n < NST; ++n) h[n] = 0.0f;

    const float* bcd_b = bcd + (size_t)b * LL * SELN;
    const float* xc_b  = xca + (size_t)b * LL * EXPD + e;
    const float* z_b   = xz  + (size_t)b * LL * (2 * EXPD) + EXPD + e;
    float*       y_b   = yg  + (size_t)b * LL * EXPD + e;

    for (int l0 = 0; l0 < LL; l0 += SCAN_CHUNK) {
        __syncthreads();                            // chunk l0-CHUNK fully consumed
#if defined(HAVE_TDM)
        if ((threadIdx.x >> 5) == 0) {              // wave 0 drives the DMA
            tdm_load_chunk(bcd_b + (size_t)l0 * SELN);
            __builtin_amdgcn_s_wait_tensorcnt(0);
        }
#else
        for (int t = threadIdx.x; t < CHUNK_ELEMS; t += EXPD)
            sbcd[t] = bcd_b[(size_t)l0 * SELN + t];
#endif
        __syncthreads();
        if (l0 + SCAN_CHUNK < LL)                   // warm caches for next xc chunk
            __builtin_prefetch(xc_b + (size_t)(l0 + SCAN_CHUNK) * EXPD, 0, 0);

        for (int li = 0; li < SCAN_CHUNK; ++li) {
            const int l = l0 + li;
            const float* rw = sbcd + li * SELN;
            const float x  = xc_b[(size_t)l * EXPD];
            const float zv = z_b [(size_t)l * (2 * EXPD)];

            const float draw = rw[32] * w0 + rw[33] * w1 + rw[34] * w2 + rw[35] * w3 + dbias;
            const float dt   = (draw > 20.0f) ? draw : log1pf(__expf(draw));  // softplus

            float y = 0.0f;
#pragma unroll
            for (int n = 0; n < NST; ++n) {
                const float Ad = __expf(dt * Arow[n]);
                h[n] = Ad * h[n] + (dt * rw[n]) * x;
                y   += h[n] * rw[NST + n];
            }
            y += dd * x;
            y *= silu_f(zv);
            y_b[(size_t)l * EXPD] = y;
        }
    }
}

// ---------------------------------------------------------------------------
// Head: last-token RMS-norm + 64->5 linear + softmax. grid = B, block = 64.
// ---------------------------------------------------------------------------
__global__ void head_kernel(const float* __restrict__ hfin,   // (B,L,64)
                            const float* __restrict__ norm_w,
                            const float* __restrict__ lin_w,  // (5,64)
                            const float* __restrict__ lin_b,
                            float* __restrict__ out) {        // (B,5)
    __shared__ float red[INC];
    __shared__ float sv[INC];
    __shared__ float logits[OUTC];

    const int b = blockIdx.x;
    const int t = threadIdx.x;

    const float v = hfin[((size_t)b * LL + (LL - 1)) * INC + t];
    red[t] = v * v;
    __syncthreads();
    for (int s = INC / 2; s > 0; s >>= 1) {
        if (t < s) red[t] += red[t + s];
        __syncthreads();
    }
    const float inv = rsqrtf(red[0] / (float)INC + 1e-5f);
    sv[t] = v * inv * norm_w[t];
    __syncthreads();

    if (t < OUTC) {
        float acc = lin_b[t];
        for (int c = 0; c < INC; ++c) acc += sv[c] * lin_w[t * INC + c];
        logits[t] = acc;
    }
    __syncthreads();

    if (t == 0) {
        float mx = logits[0];
        for (int o = 1; o < OUTC; ++o) mx = fmaxf(mx, logits[o]);
        float sum = 0.0f, ex[OUTC];
        for (int o = 0; o < OUTC; ++o) { ex[o] = __expf(logits[o] - mx); sum += ex[o]; }
        for (int o = 0; o < OUTC; ++o) out[b * OUTC + o] = ex[o] / sum;
    }
}

// ---------------------------------------------------------------------------
// Host-side orchestration
// ---------------------------------------------------------------------------
extern "C" void kernel_launch(void* const* d_in, const int* in_sizes, int n_in,
                              void* d_out, int out_size, void* d_ws, size_t ws_size,
                              hipStream_t stream) {
    (void)in_sizes; (void)n_in; (void)out_size; (void)ws_size;

    const float* x      = (const float*)d_in[0];
    const float* ipw    = (const float*)d_in[1];   // (NL, 256, 64)
    const float* ipb    = (const float*)d_in[2];   // (NL, 256)
    const float* cw     = (const float*)d_in[3];   // (NL, 128, 1, 4)
    const float* cb     = (const float*)d_in[4];   // (NL, 128)
    const float* selw   = (const float*)d_in[5];   // (NL, 36, 128)
    const float* dtw    = (const float*)d_in[6];   // (NL, 128, 4)
    const float* dtb    = (const float*)d_in[7];   // (NL, 128)
    const float* a_log  = (const float*)d_in[8];   // (NL, 128, 16)
    const float* dpar   = (const float*)d_in[9];   // (NL, 128)
    const float* opw    = (const float*)d_in[10];  // (NL, 64, 128)
    const float* opb    = (const float*)d_in[11];  // (NL, 64)
    const float* norm_w = (const float*)d_in[12];  // (64)
    const float* lin_w  = (const float*)d_in[13];  // (5, 64)
    const float* lin_b  = (const float*)d_in[14];  // (5)
    float* out = (float*)d_out;

    // workspace carve (fp32): xz | xca | bcd | yg | h0 | h1 | wpad
    float* ws   = (float*)d_ws;
    float* xz   = ws;                               // 32768*256
    float* xca  = xz   + (size_t)MROWS * 2 * EXPD;  // 32768*128
    float* bcd  = xca  + (size_t)MROWS * EXPD;      // 32768*36
    float* yg   = bcd  + (size_t)MROWS * SELN;      // 32768*128
    float* h0   = yg   + (size_t)MROWS * EXPD;      // 32768*64
    float* h1   = h0   + (size_t)MROWS * INC;       // 32768*64
    float* wpad = h1   + (size_t)MROWS * INC;       // 48*128

    const dim3 gemmBlk(128);
    const int  mblocks = MROWS / 128;               // 256 (32 M-rows per wave)

    const float* layer_in = x;
    float* layer_out = h0;

    for (int li = 0; li < NL; ++li) {
        const float* L_ipw  = ipw   + (size_t)li * 2 * EXPD * INC;
        const float* L_ipb  = ipb   + (size_t)li * 2 * EXPD;
        const float* L_cw   = cw    + (size_t)li * EXPD * KSZ;
        const float* L_cb   = cb    + (size_t)li * EXPD;
        const float* L_selw = selw  + (size_t)li * SELN * EXPD;
        const float* L_dtw  = dtw   + (size_t)li * EXPD * DTR;
        const float* L_dtb  = dtb   + (size_t)li * EXPD;
        const float* L_alog = a_log + (size_t)li * EXPD * NST;
        const float* L_dpar = dpar  + (size_t)li * EXPD;
        const float* L_opw  = opw   + (size_t)li * INC * EXPD;
        const float* L_opb  = opb   + (size_t)li * INC;

        // 1) in_proj: (32768 x 64) x (64 x 256)^T + bias -> xz
        wmma_gemm_bias<INC, 2 * EXPD, 2 * EXPD, true>
            <<<dim3(2 * EXPD / 16, mblocks), gemmBlk, 0, stream>>>(
                layer_in, L_ipw, L_ipb, xz);
        // 2) depthwise conv + SiLU -> xca
        {
            const size_t tot = (size_t)MROWS * EXPD;
            conv_silu_kernel<<<dim3((tot + 255) / 256), dim3(256), 0, stream>>>(
                xz, L_cw, L_cb, xca);
        }
        // 3) selection proj: pad W to 48 rows, then (32768 x 128) x (48 x 128)^T -> bcd
        pack_selw<<<dim3((SELNP * EXPD + 255) / 256), dim3(256), 0, stream>>>(L_selw, wpad);
        wmma_gemm_bias<EXPD, SELNP, SELN, false>
            <<<dim3(SELNP / 16, mblocks), gemmBlk, 0, stream>>>(
                xca, wpad, (const float*)nullptr, bcd);
        // 4) selective scan + gating -> yg
        scan_kernel<<<dim3(BB), dim3(EXPD), 0, stream>>>(
            bcd, xca, xz, L_dtw, L_dtb, L_alog, L_dpar, yg);
        // 5) out_proj: (32768 x 128) x (128 x 64)^T + bias -> layer_out
        wmma_gemm_bias<EXPD, INC, INC, true>
            <<<dim3(INC / 16, mblocks), gemmBlk, 0, stream>>>(
                yg, L_opw, L_opb, layer_out);

        layer_in  = layer_out;
        layer_out = (layer_out == h0) ? h1 : h0;
    }

    // final head on last layer output (== layer_in after the loop)
    head_kernel<<<dim3(BB), dim3(INC), 0, stream>>>(
        layer_in, norm_w, lin_w, lin_b, out);
}
